// Capsule_42949672960675
// MI455X (gfx1250) — matrix operand order repaired and checked by reading, not compiled, not run on hardware
//
#include <hip/hip_runtime.h>

// Problem constants (from reference)
#define NC    16
#define DC    64
#define SEQ   512
#define BATCH 64
#define HID   1024
#define OUTC  (NC * DC)        // 1024
#define MTOT  (SEQ * BATCH)    // 32768

typedef __attribute__((ext_vector_type(16))) __bf16       v16bf;
typedef __attribute__((ext_vector_type(2)))  __bf16       v2bf;
typedef __attribute__((ext_vector_type(8)))  float        v8f;
typedef __attribute__((ext_vector_type(8)))  unsigned int v8u;

#if defined(__has_builtin) && __has_builtin(__builtin_amdgcn_sched_barrier)
#define SCHED_FENCE() __builtin_amdgcn_sched_barrier(0)
#else
#define SCHED_FENCE()
#endif

// hardware f32->bf16 (RNE) packed convert: one u32 holds (x,y) as bf16 pair
__device__ __forceinline__ unsigned int pack2bf(float x, float y) {
    v2bf p;
    p[0] = (__bf16)x;
    p[1] = (__bf16)y;
    return __builtin_bit_cast(unsigned int, p);
}

// ---------------- GEMM: u = x @ W  (bf16 WMMA, f32 accumulate) ----------------
// X: (SEQ*BATCH, HID) row-major, row index = s*BATCH + b   (inputs as stored)
// W: (HID, OUTC) row-major
// U: (BATCH, NC, SEQ, DC) f32
#define BLK_M 128
#define BLK_N 64
#define BLK_K 64
#define LDA   (BLK_K + 8)   // ushort units, padded (144 B rows -> 8B-aligned pair stores)
#define LDB   (BLK_K + 8)

__global__ __launch_bounds__(256)
void capsule_proj_wmma(const float* __restrict__ X,
                       const float* __restrict__ W,
                       float* __restrict__ U)
{
    // double-buffered tiles: 2 * (128*72 + 64*72) * 2B = 54 KB of the 320 KB WGP LDS
    __shared__ unsigned short As[2][BLK_M * LDA];
    __shared__ unsigned short Bs[2][BLK_N * LDB];   // stored transposed: Bs[n][k]

    const int tid  = threadIdx.x;
    const int wave = tid >> 5;
    const int lane = tid & 31;
    const int lo   = lane & 15;
    const int hi   = lane >> 4;

    const int mBlock = blockIdx.y * BLK_M;
    const int nBlock = blockIdx.x * BLK_N;

    // stage one 128x64 A tile + 64x64 B tile (f32 -> bf16) into LDS buffer `bf`
    auto stage = [&](int bf, int k0) {
        // A: 8192 floats = 2048 float4s, 8 per thread; pair-pack -> uint2 stores
        #pragma unroll
        for (int it = 0; it < 8; ++it) {
            int i   = tid + it * 256;
            int row = i >> 4;
            int kk  = (i & 15) << 2;
            const float4 v = *(const float4*)(X + (size_t)(mBlock + row) * HID + k0 + kk);
            uint2 pq;
            pq.x = pack2bf(v.x, v.y);
            pq.y = pack2bf(v.z, v.w);
            *(uint2*)&As[bf][row * LDA + kk] = pq;
        }
        // B: 64x64 floats; each thread takes 2 K-rows x 4 columns so the (k,k+1)
        // bf16 pair for one column is a single packed convert + u32 store
        #pragma unroll
        for (int it = 0; it < 2; ++it) {
            int j = tid + it * 256;
            int k = (j >> 4) << 1;
            int c = (j & 15) << 2;
            const float4 v0 = *(const float4*)(W + (size_t)(k0 + k)     * OUTC + nBlock + c);
            const float4 v1 = *(const float4*)(W + (size_t)(k0 + k + 1) * OUTC + nBlock + c);
            *(unsigned int*)&Bs[bf][(c + 0) * LDB + k] = pack2bf(v0.x, v1.x);
            *(unsigned int*)&Bs[bf][(c + 1) * LDB + k] = pack2bf(v0.y, v1.y);
            *(unsigned int*)&Bs[bf][(c + 2) * LDB + k] = pack2bf(v0.z, v1.z);
            *(unsigned int*)&Bs[bf][(c + 3) * LDB + k] = pack2bf(v0.w, v1.w);
        }
    };

    // A fragment (16x32 bf16): lanes 0-15 K{0..7,16..23},
    // lanes 16-31 K{8..15,24..31}, 2 bf16 per VGPR
    auto load_afrag = [&](int bf, int ks) -> v16bf {
        v8u au;
        const unsigned short* arow = &As[bf][(wave * 16 + lo) * LDA + ks];
        #pragma unroll
        for (int v = 0; v < 8; ++v) {
            int kb = ((v >> 2) << 4) + (hi << 3) + ((v & 3) << 1);
            au[v] = *(const unsigned int*)(arow + kb);
        }
        return __builtin_bit_cast(v16bf, au);
    };

    // B fragment (32x16): lane column n = lo; lanes 0-15 K{0..7,16..23},
    // lanes 16-31 K{8..15,24..31}; from transposed Bs
    auto load_bfrag = [&](int bf, int nt, int ks) -> v16bf {
        v8u bu;
        const unsigned short* bcol = &Bs[bf][(nt * 16 + lo) * LDB + ks];
        #pragma unroll
        for (int v = 0; v < 8; ++v)
            bu[v] = *(const unsigned int*)(bcol + ((v >> 2) << 4) + (hi << 3) + ((v & 3) << 1));
        return __builtin_bit_cast(v16bf, bu);
    };

    v8f acc[4];
    #pragma unroll
    for (int nt = 0; nt < 4; ++nt)
        #pragma unroll
        for (int r = 0; r < 8; ++r) acc[nt][r] = 0.0f;

    stage(0, 0);
    int buf = 0;

    for (int k0 = 0; k0 < HID; k0 += BLK_K) {
        __syncthreads();                       // staging of `buf` complete
        if (k0 + BLK_K < HID)
            stage(buf ^ 1, k0 + BLK_K);        // overlap next-tile stage with WMMAs

        #pragma unroll
        for (int ks = 0; ks < BLK_K; ks += 32) {
            // preload ALL fragments, then fence the scheduler: the 10 ds_loads
            // issue as one clause ahead of the WMMAs, regalloc keeps b0..b3 in
            // distinct VGPRs, and waits become partial (dscnt 6/4/2/0) so LDS
            // latency overlaps WMMA execution.
            v16bf a  = load_afrag(buf, ks);
            v16bf b0 = load_bfrag(buf, 0, ks);
            v16bf b1 = load_bfrag(buf, 1, ks);
            v16bf b2 = load_bfrag(buf, 2, ks);
            v16bf b3 = load_bfrag(buf, 3, ks);
            SCHED_FENCE();

            acc[0] = __builtin_amdgcn_wmma_f32_16x16x32_bf16(
                false, a, false, b0, (short)0, acc[0], false, false);
            acc[1] = __builtin_amdgcn_wmma_f32_16x16x32_bf16(
                false, a, false, b1, (short)0, acc[1], false, false);
            acc[2] = __builtin_amdgcn_wmma_f32_16x16x32_bf16(
                false, a, false, b2, (short)0, acc[2], false, false);
            acc[3] = __builtin_amdgcn_wmma_f32_16x16x32_bf16(
                false, a, false, b3, (short)0, acc[3], false, false);
        }
        buf ^= 1;
    }

    // ---- store into U[b][n][s][d]; row = s*64+b, col = n*64+d ----
    #pragma unroll
    for (int nt = 0; nt < 4; ++nt) {
        int col = nBlock + nt * 16 + lo;
        int n = col >> 6, d = col & 63;
        #pragma unroll
        for (int r = 0; r < 8; ++r) {
            int row = mBlock + wave * 16 + hi * 8 + r;
            int s = row >> 6, b = row & 63;
            U[(((size_t)b * NC + n) * SEQ + s) * DC + d] = acc[nt][r];
        }
    }
}

// ---------------- Dynamic routing: one workgroup per batch element ----------------
__global__ __launch_bounds__(512)
void capsule_routing(const float* __restrict__ U, float* __restrict__ Out)
{
    __shared__ float logit[NC][SEQ + 1];           // +1 pad: break 64-bank stride
    __shared__ float cmat[NC][SEQ + 1];
    __shared__ __align__(16) float vcap[NC][DC];

    const int tid = threadIdx.x;
    const int bb  = blockIdx.x;
    const float* ub = U + (size_t)bb * NC * SEQ * DC;

    for (int i = tid; i < NC * SEQ; i += 512)
        logit[i >> 9][i & (SEQ - 1)] = 0.0f;
    __syncthreads();

    const int wv = tid >> 5;   // wave id == capsule id in stage 2
    const int ln = tid & 31;

    for (int iter = 0; iter < 4; ++iter) {
        // ---- softmax over capsules, one thread per sequence position s ----
        {
            const int s = tid;
            float m = logit[0][s];
            #pragma unroll
            for (int n = 1; n < NC; ++n) m = fmaxf(m, logit[n][s]);
            float e[NC], sum = 0.0f;
            #pragma unroll
            for (int n = 0; n < NC; ++n) { e[n] = __expf(logit[n][s] - m); sum += e[n]; }
            const float inv = 1.0f / sum;
            #pragma unroll
            for (int n = 0; n < NC; ++n) cmat[n][s] = e[n] * inv;
        }
        __syncthreads();

        // ---- v[n] = squash(sum_s c[n][s] * u[n][s][:]) ; wave wv -> capsule wv ----
        {
            const float* un = ub + (size_t)wv * SEQ * DC;
            float a0 = 0.0f, a1 = 0.0f;
            for (int s = 0; s < SEQ; ++s) {
                const float c = cmat[wv][s];          // wave-uniform broadcast
                a0 = fmaf(c, un[s * DC + ln],      a0);
                a1 = fmaf(c, un[s * DC + ln + 32], a1);
            }
            float nrm = a0 * a0 + a1 * a1;            // wave32 reduction over d=64
            #pragma unroll
            for (int off = 16; off > 0; off >>= 1)
                nrm += __shfl_xor(nrm, off, 32);
            const float inv = rsqrtf(nrm + 1e-7f);
            vcap[wv][ln]      = a0 * inv;
            vcap[wv][ln + 32] = a1 * inv;
        }
        __syncthreads();

        if (iter < 3) {
            // ---- b[n][s] += dot(v[n], u[n][s][:])  (u is L2-resident) ----
            for (int p = tid; p < NC * SEQ; p += 512) {
                const int n = p >> 9;
                const int s = p & (SEQ - 1);
                const float4* uv = (const float4*)(ub + ((size_t)n * SEQ + s) * DC);
                const float4* vv = (const float4*)(&vcap[n][0]);
                float dot = 0.0f;
                #pragma unroll
                for (int q = 0; q < DC / 4; ++q) {
                    const float4 x = uv[q], y = vv[q];
                    dot += x.x * y.x + x.y * y.y + x.z * y.z + x.w * y.w;
                }
                logit[n][s] += dot;
            }
            __syncthreads();
        }
    }

    // ---- write (B, NC, DC) output ----
    for (int i = tid; i < NC * DC; i += 512)
        Out[(size_t)bb * NC * DC + i] = vcap[i >> 6][i & 63];
}

extern "C" void kernel_launch(void* const* d_in, const int* in_sizes, int n_in,
                              void* d_out, int out_size, void* d_ws, size_t ws_size,
                              hipStream_t stream)
{
    (void)in_sizes; (void)n_in; (void)out_size; (void)ws_size;
    const float* X = (const float*)d_in[0];  // (SEQ, BATCH, HID) == (MTOT, HID) rows (s,b)
    const float* W = (const float*)d_in[1];  // (1, HID, OUTC)
    float* U = (float*)d_ws;                 // 128 MB scratch: (BATCH, NC, SEQ, DC) f32

    dim3 g1(OUTC / BLK_N, MTOT / BLK_M);     // (16, 256)
    capsule_proj_wmma<<<g1, 256, 0, stream>>>(X, W, U);
    capsule_routing<<<BATCH, 512, 0, stream>>>(U, (float*)d_out);
}
